// WeightOnlyLinear_74783970558269
// MI455X (gfx1250) — matrix operand order repaired and checked by reading, not compiled, not run on hardware
//
#include <hip/hip_runtime.h>
#include <hip/hip_bf16.h>

typedef __attribute__((ext_vector_type(16))) __bf16 v16bf;
typedef __attribute__((ext_vector_type(2)))  __bf16 v2bf;
typedef __attribute__((ext_vector_type(8)))  float  v8f;

// Problem constants (fixed by the reference).
constexpr int K_DIM = 4096;
constexpr int N_DIM = 4096;              // out_features
constexpr int M_DIM = 4 * 2048;          // batch * seq
constexpr int GROUPSIZE = 128;
constexpr int KWORDS = K_DIM / 8;        // int32 words per weight row
constexpr int KGROUPS = K_DIM / GROUPSIZE;

// Tiling.
constexpr int BM = 128, BN = 128, BK = 64;
constexpr int BKP = BK + 8;              // +16B pad per row: kills LDS bank conflicts
constexpr int WM = 64, WN = 32;          // per-wave tile
constexpr int MT = WM / 16;              // 4 m-tiles per wave
constexpr int NT = WN / 16;              // 2 n-tiles per wave

// Pack two fp32 into one dword of bf16 — lowers to v_cvt_pk_bf16_f32 on gfx1250.
__device__ __forceinline__ unsigned int pack2bf(float a, float b) {
    v2bf t;
    t.x = (__bf16)a;
    t.y = (__bf16)b;
    union { v2bf v; unsigned int u; } c;
    c.v = t;
    return c.u;
}

// Signed 4-bit field extract: v_bfe_i32 (1 VALU) instead of shl+ashr (2 VALU).
__device__ __forceinline__ int nib(int p, int e) {
#if __has_builtin(__builtin_amdgcn_sbfe)
    return __builtin_amdgcn_sbfe(p, 28 - 4 * e, 4);
#else
    return (p << (4 * e)) >> 28;
#endif
}

union V16 {
    v16bf v;
    uint4 q[2];
};

// Per-thread register staging for one K-block (double-buffer pipeline).
struct Stage {
    float4 xv[8];   // 128x64 fp32 x-tile / 256 threads
    int    wv[4];   // 128x8 int32 packed-weight words / 256 threads
    float  sv[4];   // matching group scales
};

__device__ __forceinline__ void load_stage(const float* __restrict__ x,
                                           const int*   __restrict__ pw,
                                           const float* __restrict__ scale,
                                           int blockM, int blockN, int k0, int tid,
                                           Stage& st) {
    #pragma unroll
    for (int t = 0; t < 8; ++t) {
        const int seg = tid + t * 256;
        const int row = seg >> 4;
        const int kc  = (seg & 15) << 2;
        st.xv[t] = *reinterpret_cast<const float4*>(
            x + (size_t)(blockM + row) * K_DIM + k0 + kc);
    }
    #pragma unroll
    for (int t = 0; t < 4; ++t) {
        const int w   = tid + t * 256;       // 0..1023
        const int row = w >> 3;              // 0..127
        const int wc  = w & 7;               // word within row
        const int o   = blockN + row;
        st.wv[t] = pw[(size_t)o * KWORDS + (k0 >> 3) + wc];
        st.sv[t] = scale[(size_t)o * KGROUPS + (k0 >> 7)];
    }
}

__device__ __forceinline__ void stage_to_lds(unsigned short* __restrict__ dA,
                                             unsigned short* __restrict__ dB,
                                             int tid, const Stage& st) {
    #pragma unroll
    for (int t = 0; t < 8; ++t) {
        const int seg = tid + t * 256;
        const int row = seg >> 4;
        const int kc  = (seg & 15) << 2;
        uint2 pk;
        pk.x = pack2bf(st.xv[t].x, st.xv[t].y);
        pk.y = pack2bf(st.xv[t].z, st.xv[t].w);
        *reinterpret_cast<uint2*>(&dA[row * BKP + kc]) = pk;
    }
    #pragma unroll
    for (int t = 0; t < 4; ++t) {
        const int w   = tid + t * 256;
        const int row = w >> 3;
        const int wc  = w & 7;
        const int p   = st.wv[t];
        const float s = st.sv[t];
        // element e (k = word*8 + e) = sign-extended nibble [31-4e : 28-4e]
        uint4 pk;
        pk.x = pack2bf(nib(p, 0) * s, nib(p, 1) * s);
        pk.y = pack2bf(nib(p, 2) * s, nib(p, 3) * s);
        pk.z = pack2bf(nib(p, 4) * s, nib(p, 5) * s);
        pk.w = pack2bf(nib(p, 6) * s, nib(p, 7) * s);
        *reinterpret_cast<uint4*>(&dB[row * BKP + wc * 8]) = pk;
    }
}

__global__ __launch_bounds__(256, 2)
void WeightOnlyLinear_74783970558269_kernel(const float* __restrict__ x,
                                            const int*   __restrict__ pw,
                                            const float* __restrict__ scale,
                                            const float* __restrict__ bias,
                                            float*       __restrict__ out) {
    __shared__ unsigned short sA[2][BM * BKP];   // x tile, bf16, [M][K] row-major
    __shared__ unsigned short sB[2][BN * BKP];   // dequant W tile, bf16, [N][K] row-major

    const int tid  = threadIdx.x;
    const int wave = tid >> 5;
    const int lane = tid & 31;
    const int wm   = wave >> 2;          // 0..1
    const int wn   = wave & 3;           // 0..3
    const int half = lane >> 4;          // 0..1 (lane group)
    const int lrow = lane & 15;

    const int blockN = blockIdx.x * BN;
    const int blockM = blockIdx.y * BM;

    // Accumulators seeded with bias (C/D layout: VGPR j -> row half*8+j, col lrow).
    v8f acc[MT][NT];
    #pragma unroll
    for (int tn = 0; tn < NT; ++tn) {
        const float b = bias[blockN + wn * WN + tn * 16 + lrow];
        #pragma unroll
        for (int tm = 0; tm < MT; ++tm) {
            #pragma unroll
            for (int j = 0; j < 8; ++j) acc[tm][tn][j] = b;
        }
    }

    // ---- Pipeline prologue: stage K-block 0 into LDS buffer 0.
    Stage st;
    load_stage(x, pw, scale, blockM, blockN, 0, tid, st);
    stage_to_lds(&sA[0][0], &sB[0][0], tid, st);
    __syncthreads();

    int cur = 0;
    for (int k0 = 0; k0 < K_DIM; k0 += BK) {
        const bool has_next = (k0 + BK) < K_DIM;

        // 1) Issue global loads for the NEXT K-block (latency hides under WMMAs).
        if (has_next) {
            load_stage(x, pw, scale, blockM, blockN, k0 + BK, tid, st);
        }
        // L2 pre-warm two blocks ahead (global_prefetch_b8).
        if (k0 + 2 * BK < K_DIM) {
            __builtin_prefetch(x + (size_t)(blockM + (tid >> 1)) * K_DIM + k0 + 2 * BK + (tid & 1) * 32, 0, 1);
            __builtin_prefetch(pw + (size_t)(blockN + (tid & 127)) * KWORDS + ((k0 + 2 * BK) >> 3), 0, 1);
        }

        // 2) Compute on current buffer: 2 K-steps of 32, 8 WMMAs each.
        const unsigned short* cA = &sA[cur][0];
        const unsigned short* cB = &sB[cur][0];
        #pragma unroll
        for (int ks = 0; ks < 2; ++ks) {
            const int kb = ks * 32;
            V16 afrag[MT];
            V16 bfrag[NT];
            #pragma unroll
            for (int tm = 0; tm < MT; ++tm) {
                // A layout: m = lrow, first 8 at k = kb + half*8, next 8 at +16
                const unsigned short* pa =
                    &cA[(wm * WM + tm * 16 + lrow) * BKP + kb + half * 8];
                afrag[tm].q[0] = *reinterpret_cast<const uint4*>(pa);
                afrag[tm].q[1] = *reinterpret_cast<const uint4*>(pa + 16);
            }
            #pragma unroll
            for (int tn = 0; tn < NT; ++tn) {
                // B layout: n = lrow, first 8 at k = kb + half*8, next 8 at +16
                const unsigned short* pb =
                    &cB[(wn * WN + tn * 16 + lrow) * BKP + kb + half * 8];
                bfrag[tn].q[0] = *reinterpret_cast<const uint4*>(pb);
                bfrag[tn].q[1] = *reinterpret_cast<const uint4*>(pb + 16);
            }
            #pragma unroll
            for (int tm = 0; tm < MT; ++tm) {
                #pragma unroll
                for (int tn = 0; tn < NT; ++tn) {
                    acc[tm][tn] = __builtin_amdgcn_wmma_f32_16x16x32_bf16(
                        false, afrag[tm].v, false, bfrag[tn].v,
                        (short)0, acc[tm][tn], false, false);
                }
            }
        }

        // 3) Convert + store the prefetched K-block into the other buffer.
        if (has_next) {
            stage_to_lds(&sA[cur ^ 1][0], &sB[cur ^ 1][0], tid, st);
        }

        // 4) Single barrier per iteration closes both hazards:
        //    - next iter reads LDS[cur^1] (written above by all waves)
        //    - iter after that overwrites LDS[cur] (all reads of it done here)
        __syncthreads();
        cur ^= 1;
    }

    // ---- Epilogue: scatter accumulators (bias already folded in).
    #pragma unroll
    for (int tm = 0; tm < MT; ++tm) {
        #pragma unroll
        for (int tn = 0; tn < NT; ++tn) {
            const int col   = blockN + wn * WN + tn * 16 + lrow;
            const int rbase = blockM + wm * WM + tm * 16 + half * 8;
            #pragma unroll
            for (int j = 0; j < 8; ++j) {
                out[(size_t)(rbase + j) * N_DIM + col] = acc[tm][tn][j];
            }
        }
    }
}

extern "C" void kernel_launch(void* const* d_in, const int* in_sizes, int n_in,
                              void* d_out, int out_size, void* d_ws, size_t ws_size,
                              hipStream_t stream) {
    const float* x     = (const float*)d_in[0];
    const int*   pw    = (const int*)d_in[1];
    const float* scale = (const float*)d_in[2];
    const float* bias  = (const float*)d_in[3];
    float*       out   = (float*)d_out;

    dim3 grid(N_DIM / BN, M_DIM / BM);   // (32, 64)
    dim3 block(256);
    WeightOnlyLinear_74783970558269_kernel<<<grid, block, 0, stream>>>(x, pw, scale, bias, out);
}
